// EMTransformer_7533372637378
// MI455X (gfx1250) — compile-verified, matrix operand order checked by reading, not run on hardware
//
#include <hip/hip_runtime.h>
#include <hip/hip_bf16.h>
#include <math.h>

// ---------------------------------------------------------------------------
// Problem constants (from the reference)
// ---------------------------------------------------------------------------
#define BATCH   2
#define DMODEL  128
#define NSEL    336896            // 1024 + 8192 + 65536 + 262144
#define NPAD    524288            // next pow2 >= NSEL for global bitonic argsort

// d_out float offsets (reference tuple, concatenated flat)
#define OUT_SCORES 0
#define OUT_XY     673792         // B*NSEL
#define OUT_IJ     2021376        // OUT_XY + B*NSEL*2
#define OUT_ORDER  4042752       // OUT_IJ + B*NSEL*3

// ---------------------------------------------------------------------------
// CDNA5 types
// ---------------------------------------------------------------------------
typedef __attribute__((ext_vector_type(16))) __bf16        v16bf;
typedef __attribute__((ext_vector_type(8)))  float         v8f;
typedef __attribute__((ext_vector_type(4)))  unsigned int  u32x4;
typedef __attribute__((ext_vector_type(4)))  int           i32x4;
typedef __attribute__((ext_vector_type(8)))  int           i32x8;

union Frag { v16bf v; unsigned int u[8]; };

// TDM toolchain dispatch: therock-10.0 headers ship the 6-arg builtin form.
#if __has_builtin(__builtin_amdgcn_tensor_load_to_lds)
  #define HAVE_TDM 1
  #if __has_include(<hip/amd_detail/amd_gfx1250_TDM.h>)
    #define TDM_6ARG 1
  #endif
#endif

__device__ __forceinline__ unsigned short f2bf(float f) {
    union { float f; unsigned int u; } c; c.f = f;
    unsigned int r = c.u + 0x7FFFu + ((c.u >> 16) & 1u);   // round-to-nearest-even
    return (unsigned short)(r >> 16);
}

__device__ __forceinline__ float geluf(float x) {
    const float c0 = 0.7978845608028654f, c1 = 0.044715f;
    float u = c0 * (x + c1 * x * x * x);
#if __has_builtin(__builtin_amdgcn_tanhf)
    float t = __builtin_amdgcn_tanhf(u);          // v_tanh_f32 (gfx1250 TRANS op)
#else
    float e = __expf(2.0f * u);                   // branch-free fallback
    float t = (e - 1.0f) / (e + 1.0f);
#endif
    return 0.5f * x * (1.0f + t);
}

// ---------------------------------------------------------------------------
// Fused kernel: TDM-staged feat tile + PE + upsample-gate + MLP -> score.
// 256 threads / block (8 wave32), 64 tokens / block. LDS = exactly 64 KB:
//   sW  : 32 KB  W1^T as bf16
//   buf : 32 KB  arena: [f32 feat staging] -> [bf16 A-tile | sScore | sUp]
// ---------------------------------------------------------------------------
__global__ __launch_bounds__(256)
void score_kernel(const float* __restrict__ feat, const float* __restrict__ fw,
                  const float* __restrict__ alpha, const float* __restrict__ W1,
                  const float* __restrict__ b1, const float* __restrict__ W2,
                  const float* __restrict__ b2, const float* __restrict__ prevScore,
                  const float* __restrict__ upK, float* __restrict__ outScore,
                  int H, int W, int level)
{
    __shared__ unsigned short sW[DMODEL][DMODEL];          // 32 KB
    __shared__ __align__(16) char buf[64 * DMODEL * 4];    // 32 KB arena

    float*          sFeat  = (float*)buf;                                  // phase 1
    unsigned short (*sA)[DMODEL] = (unsigned short (*)[DMODEL])buf;        // phase 3 (16 KB)
    float*          sScore = (float*)(buf + 64 * DMODEL * 2);              // +256 B
    float*          sUp    = (float*)(buf + 64 * DMODEL * 2 + 256);        // +256 B

    const int tid = threadIdx.x;
    const int b   = blockIdx.y;
    const int HW  = H * W;
    const int t0  = blockIdx.x * 64;
    const float* src = feat + ((size_t)b * HW + t0) * DMODEL;

    // ---- Phase 1: stream the 32 KB feat tile into LDS -----------------------
#if defined(HAVE_TDM)
    if (tid < 32) {                       // one wave issues the DMA (EXEC ignored)
        unsigned long long ga = (unsigned long long)(size_t)src;
        unsigned ldsOff = (unsigned)(size_t)(void*)buf;
        // D# group 0: count=1, lds_addr, 57-bit global_addr, type=2
        u32x4 g0 = { 1u, ldsOff, (unsigned)ga,
                     ((unsigned)((ga >> 32) & 0x01FFFFFFull)) | 0x80000000u };
        // D# group 1: data_size=4B, 1-D tile of 8192 elements
        i32x8 g1 = { (int)(2u << 16),          // [17:16] data_size = 4 bytes
                     (int)(8192u << 16),       // tensor_dim0[15:0] @ [63:48]
                     (int)(1u << 16),          // tensor_dim0 hi=0, tensor_dim1=1
                     (int)(8192u << 16),       // tile_dim0 = 8192 @ [127:112]
                     1,                        // tile_dim1 = 1
                     8192, 0, 0 };             // tensor_dim0_stride = 8192
        i32x4 gz = { 0, 0, 0, 0 };
  #if defined(TDM_6ARG)
        i32x8 gz8 = { 0, 0, 0, 0, 0, 0, 0, 0 };
        __builtin_amdgcn_tensor_load_to_lds(g0, g1, gz, gz, gz8, 0);
  #else
        __builtin_amdgcn_tensor_load_to_lds(g0, g1, gz, gz, 0);
  #endif
  #if __has_builtin(__builtin_amdgcn_s_wait_tensorcnt)
        __builtin_amdgcn_s_wait_tensorcnt(0);
  #endif
    }
#else
    {   // fallback: vectorized cooperative copy
        const float4* s4 = (const float4*)src;
        float4* d4 = (float4*)sFeat;
        for (int e = tid; e < 64 * DMODEL / 4; e += 256) d4[e] = s4[e];
    }
#endif

    // W1^T -> LDS as bf16 while the DMA streams (W1 is L2-resident)
    __builtin_prefetch(W1 + (tid << 6), 0, 0);
    for (int e = tid; e < DMODEL * DMODEL; e += 256) {
        int n = e & 127, k = e >> 7;
        sW[n][k] = f2bf(W1[k * DMODEL + n]);
    }
    __syncthreads();

    // ---- Phase 2: drain staging to registers --------------------------------
    // Thread owns one feature dim dFix for 32 tokens (t = tBase + 2s).
    const int dFix  = tid & 127;
    const int tBase = tid >> 7;
    float regs[32];
    #pragma unroll
    for (int s = 0; s < 32; ++s)
        regs[s] = sFeat[(tBase + 2 * s) * DMODEL + dFix];

    // per-token transposed-conv upsample gate (threads 0..63)
    float upReg = 0.0f;
    if (tid < 64 && level > 0) {
        int tok = t0 + tid;
        int i = tok / W, j = tok - (tok / W) * W;
        int Hp = H >> 1, Wp = W >> 1;
        const float* ps = prevScore + (size_t)b * Hp * Wp;
        #pragma unroll
        for (int ki = 0; ki < 3; ++ki)
            #pragma unroll
            for (int kj = 0; kj < 3; ++kj) {
                int i2 = i + 1 - ki, j2 = j + 1 - kj;
                if (!(i2 & 1) && !(j2 & 1)) {
                    int ii = i2 >> 1, jj = j2 >> 1;
                    if (ii >= 0 && ii < Hp && jj >= 0 && jj < Wp)
                        upReg += ps[ii * Wp + jj] * upK[ki * 3 + kj];
                }
            }
        upReg *= alpha[level - 1];
    }
    __syncthreads();                       // staging reads complete

    if (tid < 64) { sUp[tid] = upReg; sScore[tid] = 0.0f; }
    __syncthreads();

    // ---- Phase 3: x = (feat + fourier_pe) * (1 + alpha*up) -> bf16 A-tile ---
    const float TWO_PI = 6.283185307179586f;
    const int   c   = dFix & 63;
    const float fwx = fw[c], fwy = fw[64 + c], fwz = fw[128 + c];
    #pragma unroll
    for (int s = 0; s < 32; ++s) {
        int t   = tBase + 2 * s;
        int tok = t0 + t;
        int i = tok / W, j = tok - (tok / W) * W;
        float xn = (j + 0.5f) / (float)W, yn = (i + 0.5f) / (float)H;
        float proj = TWO_PI * (xn * fwx + yn * fwy + (float)level * fwz);
        float pe = (dFix < 64) ? __sinf(proj) : __cosf(proj);
        sA[t][dFix] = f2bf((regs[s] + pe) * (1.0f + sUp[t]));
    }
    __syncthreads();

    // ---- Phase 4: WMMA GEMM + fused gelu/W2 epilogue ------------------------
    // wave tiling: M-subtile = wave>>1, N-subtiles = (wave&1)*64 + {0,16,32,48}
    const int lane  = tid & 31;
    const int wave  = tid >> 5;
    const int msub  = wave >> 1;
    const int nbase = (wave & 1) * 64;
    const int half  = lane >> 4;
    const int lm    = lane & 15;

    // A fragments, ISA 16-bit A layout: VGPR p holds K pair {base, base+1};
    // lanes 16-31 offset K by +8; p>=4 -> +16.
    Frag afrag[4];
    #pragma unroll
    for (int kc = 0; kc < 4; ++kc)
        #pragma unroll
        for (int p = 0; p < 8; ++p) {
            int kk = kc * 32 + ((p & 4) ? 16 : 0) + (p & 3) * 2 + half * 8;
            afrag[kc].u[p] = *(const unsigned int*)&sA[msub * 16 + lm][kk];
        }

    #pragma unroll
    for (int nt = 0; nt < 4; ++nt) {
        int n0 = nbase + nt * 16;
        v8f acc = {};
        #pragma unroll
        for (int kc = 0; kc < 4; ++kc) {
            Frag bfrag;
            #pragma unroll
            for (int p = 0; p < 8; ++p) {
                int kk = kc * 32 + ((p & 4) ? 16 : 0) + (p & 3) * 2 + half * 8;
                bfrag.u[p] = *(const unsigned int*)&sW[n0 + lm][kk];
            }
            acc = __builtin_amdgcn_wmma_f32_16x16x32_bf16(
                      false, afrag[kc].v, false, bfrag.v,
                      (short)0, acc, false, false);
        }
        // fused epilogue: +b1 -> gelu -> *W2, reduce over n via LDS atomics
        int n = n0 + lm;
        float b1n = b1[n], w2n = W2[n];
        #pragma unroll
        for (int r = 0; r < 8; ++r) {
            int m = r + (half ? 8 : 0);
            float h = geluf(acc[r] + b1n);
            atomicAdd(&sScore[msub * 16 + m], h * w2n);
        }
    }
    __syncthreads();

    if (tid < 64)
        outScore[(size_t)b * HW + t0 + tid] = sScore[tid] + b2[0];
}

// ---------------------------------------------------------------------------
// Sorting / selection kernels (per-batch bitonic, key=f32 desc, payload=i32)
// ---------------------------------------------------------------------------
__global__ void init_pairs(const float* __restrict__ src, float* __restrict__ key,
                           int* __restrict__ idx, int n)
{
    int t = blockIdx.x * blockDim.x + threadIdx.x;
    if (t >= BATCH * n) return;
    key[t] = src[t];
    idx[t] = t % n;
}

__global__ void bitonic_step(float* __restrict__ key, int* __restrict__ idx,
                             int n, int j, int k)
{
    int t = blockIdx.x * blockDim.x + threadIdx.x;
    if (t >= BATCH * n) return;
    int b = t / n, i = t - b * n;
    int ixj = i ^ j;
    if (ixj <= i) return;
    float* kb = key + (size_t)b * n;
    int*   ib = idx + (size_t)b * n;
    float a = kb[i], c = kb[ixj];
    bool descSeg = ((i & k) == 0);                // descending overall
    if (descSeg ? (a < c) : (a > c)) {
        kb[i] = c; kb[ixj] = a;
        int ta = ib[i]; ib[i] = ib[ixj]; ib[ixj] = ta;
    }
}

__global__ void init_global(float* __restrict__ gKey, int* __restrict__ gIdx)
{
    int t = blockIdx.x * blockDim.x + threadIdx.x;
    if (t >= BATCH * NPAD) return;
    gKey[t] = -1e30f;
    gIdx[t] = 0;
}

__global__ void write_sel(const float* __restrict__ key, const int* __restrict__ idx,
                          int n, int k, int H, int W, int level, int selOff,
                          float* __restrict__ out, float* __restrict__ gKey,
                          int* __restrict__ gIdx)
{
    int t = blockIdx.x * blockDim.x + threadIdx.x;
    if (t >= BATCH * k) return;
    int b = t / k, r = t - b * k;
    float s = key[(size_t)b * n + r];
    int  id = idx[(size_t)b * n + r];
    int  p  = selOff + r;
    int  i  = id / W, j = id - (id / W) * W;

    out[OUT_SCORES + (size_t)b * NSEL + p] = s;
    float* xy = out + OUT_XY + ((size_t)b * NSEL + p) * 2;
    xy[0] = (j + 0.5f) / (float)W;
    xy[1] = (i + 0.5f) / (float)H;
    float* ij = out + OUT_IJ + ((size_t)b * NSEL + p) * 3;
    ij[0] = (float)i; ij[1] = (float)j; ij[2] = (float)level;

    gKey[(size_t)b * NPAD + p] = s;
    gIdx[(size_t)b * NPAD + p] = p;
}

__global__ void write_order(const int* __restrict__ gIdx, float* __restrict__ out)
{
    int t = blockIdx.x * blockDim.x + threadIdx.x;
    if (t >= BATCH * NSEL) return;
    int b = t / NSEL, r = t - b * NSEL;
    float v = (float)gIdx[(size_t)b * NPAD + r];
    out[OUT_ORDER + (size_t)b * NSEL + r] = v;

    const int subOff[6] = {4716544, 5390336, 5929368, 6333642, 6737916, 7007432};
    const int subLen[6] = {336896, 269516, 202137, 202137, 134758, 67379};
    #pragma unroll
    for (int s = 0; s < 6; ++s)
        if (r < subLen[s])
            out[subOff[s] + (size_t)b * subLen[s] + r] = v;
}

// ---------------------------------------------------------------------------
// Host-side orchestration
// ---------------------------------------------------------------------------
static inline void run_bitonic(float* key, int* idx, int n, hipStream_t stream)
{
    int threads = BATCH * n;
    int blocks  = (threads + 255) / 256;
    for (int k = 2; k <= n; k <<= 1)
        for (int j = k >> 1; j > 0; j >>= 1)
            bitonic_step<<<blocks, 256, 0, stream>>>(key, idx, n, j, k);
}

extern "C" void kernel_launch(void* const* d_in, const int* in_sizes, int n_in,
                              void* d_out, int out_size, void* d_ws, size_t ws_size,
                              hipStream_t stream)
{
    (void)in_sizes; (void)n_in; (void)out_size; (void)ws_size;

    const float* feats[4] = {(const float*)d_in[0], (const float*)d_in[1],
                             (const float*)d_in[2], (const float*)d_in[3]};
    const float* fw    = (const float*)d_in[4];
    const float* alpha = (const float*)d_in[5];
    const float* W1    = (const float*)d_in[6];
    const float* b1    = (const float*)d_in[7];
    const float* W2    = (const float*)d_in[8];
    const float* b2    = (const float*)d_in[9];
    const float* upk[3] = {(const float*)d_in[10], (const float*)d_in[11],
                           (const float*)d_in[12]};
    float* out = (float*)d_out;

    // workspace carve-up
    float* gScore = (float*)d_ws;                 // 696320 f
    float* keyBuf = gScore + 696320;              // 524288 f (max level)
    float* gKey   = keyBuf + 524288;              // BATCH*NPAD f
    int*   idxBuf = (int*)(gKey + (size_t)BATCH * NPAD);
    int*   gIdx   = idxBuf + 524288;              // BATCH*NPAD i

    const int Hs[4]  = {64, 128, 256, 512};
    const int Ws_[4] = {64, 128, 256, 512};
    const int ks[4]  = {1024, 8192, 65536, 262144};
    const int selOff[4] = {0, 1024, 9216, 74752};
    size_t lvlBase[4];
    size_t acc = 0;
    for (int l = 0; l < 4; ++l) { lvlBase[l] = acc; acc += (size_t)BATCH * Hs[l] * Ws_[l]; }

    init_global<<<(BATCH * NPAD + 255) / 256, 256, 0, stream>>>(gKey, gIdx);

    for (int l = 0; l < 4; ++l) {
        int H = Hs[l], W = Ws_[l], HW = H * W;
        dim3 grid(HW / 64, BATCH);
        score_kernel<<<grid, 256, 0, stream>>>(
            feats[l], fw, alpha, W1, b1, W2, b2,
            l ? gScore + lvlBase[l - 1] : nullptr,
            l ? upk[l - 1] : nullptr,
            gScore + lvlBase[l], H, W, l);

        init_pairs<<<(BATCH * HW + 255) / 256, 256, 0, stream>>>(
            gScore + lvlBase[l], keyBuf, idxBuf, HW);
        run_bitonic(keyBuf, idxBuf, HW, stream);

        write_sel<<<(BATCH * ks[l] + 255) / 256, 256, 0, stream>>>(
            keyBuf, idxBuf, HW, ks[l], H, W, l, selOff[l], out, gKey, gIdx);
    }

    // global argsort (descending) over concatenated selected scores
    run_bitonic(gKey, gIdx, NPAD, stream);

    write_order<<<(BATCH * NSEL + 255) / 256, 256, 0, stream>>>(gIdx, out);
}
